// DenseUNet_70884140253245
// MI455X (gfx1250) — matrix operand order repaired and checked by reading, not compiled, not run on hardware
//
#include <hip/hip_runtime.h>
#include <hip/hip_bf16.h>

// ---------------------------------------------------------------------------
// CDNA5 (gfx1250, wave32) sparse-voxel U-Net.
// Heavy 27-tap gather-GEMM convs run on v_wmma_f32_16x16x32_bf16.
// ---------------------------------------------------------------------------

typedef __attribute__((ext_vector_type(16))) __bf16 v16bf;
typedef __attribute__((ext_vector_type(8)))  float  v8f;

#define EPSV 1e-4f

// ---------------- generic helpers ----------------

__global__ void k_zero(float* __restrict__ p, long cnt) {
    long t = (long)blockIdx.x * blockDim.x + threadIdx.x;
    if (t < cnt) p[t] = 0.f;
}

// cast W[27][Ci][Co] (f32, row-major) -> Wt[27][Co][Ci] (bf16, co-major)
__global__ void k_cast_w(const float* __restrict__ W, __bf16* __restrict__ Wt,
                         int Ci, int Co) {
    long t = (long)blockIdx.x * blockDim.x + threadIdx.x;
    long total = 27L * Ci * Co;
    if (t >= total) return;
    int k  = (int)(t / (Ci * Co));
    int r  = (int)(t % (Ci * Co));
    int ci = r / Co;
    int co = r % Co;
    Wt[((long)k * Co + co) * Ci + ci] = (__bf16)W[t];
}

__global__ void k_cast_f32_bf16(const float* __restrict__ in, __bf16* __restrict__ out,
                                long cnt) {
    long t = (long)blockIdx.x * blockDim.x + threadIdx.x;
    if (t < cnt) out[t] = (__bf16)in[t];
}

// ---------------- input conv: feats (N,3) -> xv bf16 (N,32) ----------------

__global__ void k_subconv_in(const float* __restrict__ feats,
                             const int* __restrict__ nbr,
                             const float* __restrict__ Wsub,
                             __bf16* __restrict__ xv, int N) {
    long t = (long)blockIdx.x * blockDim.x + threadIdx.x;
    if (t >= (long)N * 32) return;
    int n  = (int)(t >> 5);
    int co = (int)(t & 31);
    float s = 0.f;
    #pragma unroll 1
    for (int k = 0; k < 27; ++k) {
        int idx = nbr[(long)n * 27 + k];
        if (idx >= 0) {
            const float* f = feats + (long)idx * 3;
            const float* w = Wsub + (long)k * 96 + co;   // (k*3+c)*32+co
            s += f[0] * w[0] + f[1] * w[32] + f[2] * w[64];
        }
    }
    xv[(long)n * 32 + co] = (__bf16)s;
}

// ---------------- WMMA gather-GEMM subconv ----------------
// out[n,co] = sum_k sum_ci in_bf16[nbr[n,k], ci] * Wt[k][co][ci]
// One wave handles a 16-row tile and ALL Cout tiles (A reused across B tiles).

template<int CIN, int COUT>
__global__ void __launch_bounds__(256)
k_subconv_wmma(const __bf16* __restrict__ in, const int* __restrict__ nbr,
               const __bf16* __restrict__ wt, float* __restrict__ out, int nrows) {
    const int lane = threadIdx.x & 31;
    const int wave = threadIdx.x >> 5;
    const int rowTiles = (nrows + 15) >> 4;
    int rowTile = blockIdx.x * 8 + wave;
    if (rowTile >= rowTiles) return;

    const int m      = lane & 15;
    const int rowIdx = rowTile * 16 + m;
    const bool rowOk = rowIdx < nrows;
    const int off0   = (lane < 16) ? 0 : 8;    // A: K chunk base (ISA 7.12.2)
    const int ciOff  = (lane < 16) ? 0 : 8;    // B: K chunk base (same as A)

    constexpr int CT = COUT / 16;
    v8f c[CT];
    #pragma unroll
    for (int i = 0; i < CT; ++i) c[i] = v8f{};

    #pragma unroll 1
    for (int k = 0; k < 27; ++k) {
        int idx = rowOk ? nbr[(long)rowIdx * 27 + k] : -1;
        #pragma unroll
        for (int kb = 0; kb < CIN / 32; ++kb) {
            union { v16bf v; uint4 q[2]; } a;
            if (idx >= 0) {
                const uint4* ap =
                    (const uint4*)(in + (long)idx * CIN + kb * 32 + off0);
                a.q[0] = ap[0];       // K = off0 .. off0+7
                a.q[1] = ap[2];       // K = off0+16 .. off0+23
            } else {
                a.q[0] = make_uint4(0u, 0u, 0u, 0u);
                a.q[1] = make_uint4(0u, 0u, 0u, 0u);
            }
            #pragma unroll
            for (int ct = 0; ct < CT; ++ct) {
                union { v16bf v; uint4 q[2]; } b;
                const int co = ct * 16 + m;
                const uint4* bp =
                    (const uint4*)(wt + ((long)k * COUT + co) * CIN + kb * 32 + ciOff);
                b.q[0] = bp[0];
                b.q[1] = bp[2];
                c[ct] = __builtin_amdgcn_wmma_f32_16x16x32_bf16(
                    false, a.v, false, b.v, (short)0, c[ct], false, false);
            }
        }
    }

    const int mBase = (lane < 16) ? 0 : 8;     // D: lane holds col m, rows mBase..mBase+7
    #pragma unroll
    for (int ct = 0; ct < CT; ++ct) {
        #pragma unroll
        for (int r = 0; r < 8; ++r) {
            int row = rowTile * 16 + mBase + r;
            if (row < nrows)
                out[(long)row * COUT + ct * 16 + m] = c[ct][r];
        }
    }
}

// ---------------- batch-norm ----------------

template<int C>
__global__ void k_bn_stats(const float* __restrict__ x, int n, float* __restrict__ acc) {
    __shared__ float s1[256], s2[256];
    const int tid  = threadIdx.x;
    const int c    = tid % C;
    const int rsub = tid / C;
    const int RPB  = 256 / C;
    float a1 = 0.f, a2 = 0.f;
    for (long r = (long)blockIdx.x * RPB + rsub; r < n; r += (long)gridDim.x * RPB) {
        float v = x[r * C + c];
        a1 += v;
        a2 += v * v;
    }
    s1[tid] = a1; s2[tid] = a2;
    __syncthreads();
    for (int half = RPB / 2; half >= 1; half >>= 1) {
        if (rsub < half) {
            s1[tid] += s1[tid + half * C];
            s2[tid] += s2[tid + half * C];
        }
        __syncthreads();
    }
    if (tid < C) {
        atomicAdd(&acc[c], s1[c]);
        atomicAdd(&acc[C + c], s2[c]);
    }
}

__global__ void k_bn_finalize(const float* __restrict__ acc,
                              const float* __restrict__ g, const float* __restrict__ b,
                              float* __restrict__ sb, int C, float invn) {
    int c = threadIdx.x;
    if (c < C) {
        float mu  = acc[c] * invn;
        float var = acc[C + c] * invn - mu * mu;
        float sc  = g[c] * rsqrtf(var + EPSV);
        sb[c]     = sc;
        sb[C + c] = b[c] - mu * sc;
    }
}

// relu(x*scale+bias) -> bf16 at out[r*ostride + ooff + c]
__global__ void k_bn_apply_bf16(const float* __restrict__ x, const float* __restrict__ sb,
                                __bf16* __restrict__ out, long n, int C,
                                int ostride, int ooff) {
    long t = (long)blockIdx.x * blockDim.x + threadIdx.x;
    if (t >= n * C) return;
    long r = t / C;
    int  c = (int)(t % C);
    float v = fmaxf(x[r * C + c] * sb[c] + sb[C + c], 0.f);
    out[r * ostride + ooff + c] = (__bf16)v;
}

// ---------------- down / up einsums ----------------

// down[parent[n], d] += sum_c e0[n,c] * W_down[off[n]][c][d]   (d = 0..63)
__global__ void k_down(const __bf16* __restrict__ dec_in,   // (N,64), e0 in cols 0..31
                       const int* __restrict__ parent, const int* __restrict__ coff,
                       const float* __restrict__ Wdown, float* __restrict__ down) {
    __shared__ float e[32];
    int n = blockIdx.x;
    int d = threadIdx.x;
    if (d < 32) e[d] = (float)dec_in[(long)n * 64 + d];
    __syncthreads();
    int off = coff[n];
    int p   = parent[n];
    const float* w = Wdown + (long)off * 32 * 64 + d;
    float s = 0.f;
    #pragma unroll
    for (int c = 0; c < 32; ++c) s += e[c] * w[(long)c * 64];
    atomicAdd(&down[(long)p * 64 + d], s);
}

// dec_in[n, 32+d] = sum_c e1[parent[n], c] * W_up[off[n]][c][d]   (d = 0..31)
__global__ void k_up(const __bf16* __restrict__ e1, const int* __restrict__ parent,
                     const int* __restrict__ coff, const float* __restrict__ Wup,
                     __bf16* __restrict__ dec_in) {
    __shared__ float e[64];
    int n = blockIdx.x;
    int d = threadIdx.x;                   // 0..31
    long p = parent[n];
    e[d]      = (float)e1[p * 64 + d];
    e[d + 32] = (float)e1[p * 64 + d + 32];
    __syncthreads();
    int off = coff[n];
    const float* w = Wup + (long)off * 64 * 32 + d;
    float s = 0.f;
    #pragma unroll
    for (int c = 0; c < 64; ++c) s += e[c] * w[(long)c * 32];
    dec_in[(long)n * 64 + 32 + d] = (__bf16)s;
}

// ---------------- final BN + linear head ----------------

__global__ void k_final(const float* __restrict__ t2, const float* __restrict__ sb,
                        const float* __restrict__ Wlin, const float* __restrict__ blin,
                        float* __restrict__ y, float* __restrict__ feat) {
    __shared__ float f[32];
    int n = blockIdx.x;
    int c = threadIdx.x;                   // 0..31 (one wave)
    float v = fmaxf(t2[(long)n * 32 + c] * sb[c] + sb[32 + c], 0.f);
    feat[(long)n * 32 + c] = v;
    f[c] = v;
    __syncthreads();
    if (c < 20) {
        float s = blin[c];
        #pragma unroll
        for (int i = 0; i < 32; ++i) s += f[i] * Wlin[(long)i * 20 + c];
        y[(long)n * 20 + c] = s;
    }
}

// ---------------------------------------------------------------------------

extern "C" void kernel_launch(void* const* d_in, const int* in_sizes, int n_in,
                              void* d_out, int out_size, void* d_ws, size_t ws_size,
                              hipStream_t stream) {
    const float* feats      = (const float*)d_in[0];
    const int*   nbr_fine   = (const int*)  d_in[1];
    const int*   nbr_coarse = (const int*)  d_in[2];
    const int*   parent_id  = (const int*)  d_in[3];
    const int*   child_off  = (const int*)  d_in[4];
    const float* W_sub      = (const float*)d_in[5];
    const float* W_e0       = (const float*)d_in[6];
    const float* g0         = (const float*)d_in[7];
    const float* b0         = (const float*)d_in[8];
    const float* W_down     = (const float*)d_in[9];
    const float* W_e1       = (const float*)d_in[10];
    const float* g1         = (const float*)d_in[11];
    const float* b1         = (const float*)d_in[12];
    const float* W_up       = (const float*)d_in[13];
    const float* W_dec      = (const float*)d_in[14];
    const float* g_out      = (const float*)d_in[15];
    const float* b_out      = (const float*)d_in[16];
    const float* W_lin      = (const float*)d_in[17];
    const float* b_lin      = (const float*)d_in[18];

    const int N = in_sizes[3];           // parent_id has N entries
    const int M = in_sizes[2] / 27;      // nbr_coarse rows

    // ---- workspace carve-out (256B aligned) ----
    char*  base = (char*)d_ws;
    size_t off  = 0;
    auto alloc = [&](size_t bytes) -> char* {
        char* p = base + off;
        off = (off + bytes + 255) & ~(size_t)255;
        return p;
    };
    float*  acc      = (float*) alloc(128 * sizeof(float));          // BN sums (s1|s2)
    float*  sb       = (float*) alloc(128 * sizeof(float));          // scale|bias
    __bf16* We0_t    = (__bf16*)alloc(27L * 32 * 32 * 2);
    __bf16* We1_t    = (__bf16*)alloc(27L * 64 * 64 * 2);
    __bf16* Wdec_t   = (__bf16*)alloc(27L * 64 * 32 * 2);
    __bf16* xv       = (__bf16*)alloc((size_t)N * 32 * 2);           // input-conv out
    float*  t_fine   = (float*) alloc((size_t)N * 32 * 4);           // t0 / reused as t2
    __bf16* dec_in   = (__bf16*)alloc((size_t)N * 64 * 2);           // [e0 | up] bf16
    float*  down_f   = (float*) alloc((size_t)M * 64 * 4);
    __bf16* down_bf  = (__bf16*)alloc((size_t)M * 64 * 2);
    float*  t1       = (float*) alloc((size_t)M * 64 * 4);
    __bf16* e1_bf    = (__bf16*)down_f;   // overlap: down_f dead once down_bf cast done
    (void)ws_size; (void)n_in; (void)out_size;

    float* y_out    = (float*)d_out;                 // (N,20)
    float* feat_out = (float*)d_out + (size_t)N * 20; // (N,32)

    const int TB = 256;
    auto blocks = [](long n, int t) { return (int)((n + t - 1) / t); };

    // 1) weight casts/transposes into WMMA B layout
    k_cast_w<<<blocks(27L * 32 * 32, TB), TB, 0, stream>>>(W_e0,  We0_t,  32, 32);
    k_cast_w<<<blocks(27L * 64 * 64, TB), TB, 0, stream>>>(W_e1,  We1_t,  64, 64);
    k_cast_w<<<blocks(27L * 64 * 32, TB), TB, 0, stream>>>(W_dec, Wdec_t, 64, 32);

    // 2) input conv (3->32), write bf16
    k_subconv_in<<<blocks((long)N * 32, TB), TB, 0, stream>>>(feats, nbr_fine, W_sub, xv, N);

    // 3) e0 conv (WMMA 32->32)
    {
        int rowTiles = (N + 15) / 16;
        k_subconv_wmma<32, 32><<<blocks(rowTiles, 8), 256, 0, stream>>>(
            xv, nbr_fine, We0_t, t_fine, N);
    }

    // 4) BN0 -> e0 bf16 into dec_in[:, 0:32]
    k_zero<<<1, 128, 0, stream>>>(acc, 128);
    k_bn_stats<32><<<512, 256, 0, stream>>>(t_fine, N, acc);
    k_bn_finalize<<<1, 32, 0, stream>>>(acc, g0, b0, sb, 32, 1.f / (float)N);
    k_bn_apply_bf16<<<blocks((long)N * 32, TB), TB, 0, stream>>>(
        t_fine, sb, dec_in, N, 32, 64, 0);

    // 5) down projection + segment sum (atomic f32)
    k_zero<<<blocks((long)M * 64, TB), TB, 0, stream>>>(down_f, (long)M * 64);
    k_down<<<N, 64, 0, stream>>>(dec_in, parent_id, child_off, W_down, down_f);
    k_cast_f32_bf16<<<blocks((long)M * 64, TB), TB, 0, stream>>>(
        down_f, down_bf, (long)M * 64);

    // 6) e1 conv (WMMA 64->64) on coarse grid
    {
        int rowTiles = (M + 15) / 16;
        k_subconv_wmma<64, 64><<<blocks(rowTiles, 8), 256, 0, stream>>>(
            down_bf, nbr_coarse, We1_t, t1, M);
    }

    // 7) BN1 -> e1 bf16 (overlaps dead down_f region)
    k_zero<<<1, 128, 0, stream>>>(acc, 128);
    k_bn_stats<64><<<512, 256, 0, stream>>>(t1, M, acc);
    k_bn_finalize<<<1, 64, 0, stream>>>(acc, g1, b1, sb, 64, 1.f / (float)M);
    k_bn_apply_bf16<<<blocks((long)M * 64, TB), TB, 0, stream>>>(
        t1, sb, e1_bf, M, 64, 64, 0);

    // 8) up projection into dec_in[:, 32:64]
    k_up<<<N, 32, 0, stream>>>(e1_bf, parent_id, child_off, W_up, dec_in);

    // 9) decoder conv (WMMA 64->32), reuse t_fine as t2
    {
        int rowTiles = (N + 15) / 16;
        k_subconv_wmma<64, 32><<<blocks(rowTiles, 8), 256, 0, stream>>>(
            dec_in, nbr_fine, Wdec_t, t_fine, N);
    }

    // 10) final BN + linear head
    k_zero<<<1, 128, 0, stream>>>(acc, 128);
    k_bn_stats<32><<<512, 256, 0, stream>>>(t_fine, N, acc);
    k_bn_finalize<<<1, 32, 0, stream>>>(acc, g_out, b_out, sb, 32, 1.f / (float)N);
    k_final<<<N, 32, 0, stream>>>(t_fine, sb, W_lin, b_lin, y_out, feat_out);
}
